// Attention_33251636806121
// MI455X (gfx1250) — compile-verified, matrix operand order checked by reading, not run on hardware
//
#include <hip/hip_runtime.h>

// ---------------------------------------------------------------------------
// MI455X (gfx1250) attention block:
//   LN -> QKV gemm -> causal softmax attention -> o_proj gemm + residual
// All GEMMs: V_WMMA_F32_16X16X32_BF16 (wave32), fp32 accumulate,
// double-buffered GLOBAL_LOAD_ASYNC_TO_LDS_B128 staging (ASYNCcnt pipeline).
// ---------------------------------------------------------------------------

typedef __attribute__((ext_vector_type(16))) __bf16 v16bf;
typedef __attribute__((ext_vector_type(8)))  __bf16 v8bf;
typedef __attribute__((ext_vector_type(8)))  float  v8f;

#define HIDDEN 2048
#define KEY    2048
#define BATCH  4
#define SEQ    2048
#define NROWS  (BATCH * SEQ)   // 8192
#define QKV3   (3 * KEY)       // 6144

#define BM 128
#define BN 128
#define BK 32
#define LDSPAD 40                      // padded LDS row stride (bf16), 80B
#define TILE   (128 * LDSPAD)          // elems per LDS tile buffer
#define TILEB  (TILE * 2)              // bytes per LDS tile buffer (10240)

// ---------------- bf16 helpers (round-to-nearest-even) ----------------------
static __device__ __forceinline__ unsigned short f2bf_bits(float f) {
  unsigned u = __builtin_bit_cast(unsigned, f);
  unsigned r = u + 0x7fffu + ((u >> 16) & 1u);
  if ((u & 0x7f800000u) == 0x7f800000u) r = u;   // inf/nan: truncate
  return (unsigned short)(r >> 16);
}
static __device__ __forceinline__ __bf16 f2bf(float f) {
  return __builtin_bit_cast(__bf16, f2bf_bits(f));
}
static __device__ __forceinline__ __bf16 to_bf16(float f)  { return f2bf(f); }
static __device__ __forceinline__ __bf16 to_bf16(__bf16 b) { return b; }

union V16U { v16bf v; v8bf h[2]; };

// Load one 16x32 bf16 WMMA fragment row from an LDS tile.
// A-layout: lanes 0-15 hold K={0..7,16..23}, lanes 16-31 K={8..15,24..31}.
static __device__ __forceinline__ v16bf load_frag(const __bf16* rowbase, int fh) {
  V16U t;
  t.h[0] = *(const v8bf*)(rowbase + fh * 8);
  t.h[1] = *(const v8bf*)(rowbase + 16 + fh * 8);
  return t.v;
}

// 32B async DMA global -> LDS (2 x b128). offset: applies to BOTH addresses.
// Tracked by ASYNCcnt (+2 per wave per call).
static __device__ __forceinline__ void async_copy32B(unsigned lds_off,
                                                     const __bf16* g) {
  asm volatile("global_load_async_to_lds_b128 %0, %1, off\n\t"
               "global_load_async_to_lds_b128 %0, %1, off offset:16"
               :: "v"(lds_off), "v"(g) : "memory");
}
static __device__ __forceinline__ void wait_async_le4() {
  asm volatile("s_wait_asynccnt 0x4" ::: "memory");
}
static __device__ __forceinline__ void wait_async_0() {
  asm volatile("s_wait_asynccnt 0x0" ::: "memory");
}

// ---------------- LDS-tiled transpose (+cast) to bf16 ------------------------
// src is rows x cols (leading dim ld_src); dst[c][r] (leading dim ld_dst).
template <typename T>
__global__ void transpose_to_bf16(const T* __restrict__ src, int ld_src,
                                  __bf16* __restrict__ dst, int ld_dst) {
  __shared__ __bf16 tile[64][65];
  const int c0 = blockIdx.x * 64, r0 = blockIdx.y * 64;
  const int tc = threadIdx.x & 63, tr = threadIdx.x >> 6;  // 0..3
#pragma unroll
  for (int i = 0; i < 16; i++) {
    const int r = tr + i * 4;
    tile[r][tc] = to_bf16(src[(size_t)(r0 + r) * ld_src + c0 + tc]);
  }
  __syncthreads();
#pragma unroll
  for (int i = 0; i < 16; i++) {
    const int r = tr + i * 4;
    dst[(size_t)(c0 + r) * ld_dst + r0 + tc] = tile[tc][r];
  }
}

// ---------------- LayerNorm (fp32 in, bf16 out) ------------------------------
__global__ void layernorm_bf16(const float* __restrict__ x,
                               const float* __restrict__ gamma,
                               const float* __restrict__ beta,
                               __bf16* __restrict__ out) {
  __shared__ float red[256];
  const int row = blockIdx.x;
  const int tid = threadIdx.x;
  const float* xr = x + (size_t)row * HIDDEN;

  float v[8];
  float s = 0.f;
#pragma unroll
  for (int i = 0; i < 8; i++) { v[i] = xr[tid + i * 256]; s += v[i]; }
  red[tid] = s; __syncthreads();
  for (int o = 128; o > 0; o >>= 1) {
    if (tid < o) red[tid] += red[tid + o];
    __syncthreads();
  }
  const float mu = red[0] * (1.0f / HIDDEN);
  __syncthreads();

  float ss = 0.f;
#pragma unroll
  for (int i = 0; i < 8; i++) { float d = v[i] - mu; ss += d * d; }
  red[tid] = ss; __syncthreads();
  for (int o = 128; o > 0; o >>= 1) {
    if (tid < o) red[tid] += red[tid + o];
    __syncthreads();
  }
  const float rstd = rsqrtf(red[0] * (1.0f / HIDDEN) + 1e-5f);

  __bf16* orow = out + (size_t)row * HIDDEN;
#pragma unroll
  for (int i = 0; i < 8; i++) {
    const int c = tid + i * 256;
    orow[c] = f2bf((v[i] - mu) * rstd * gamma[c] + beta[c]);
  }
}

// ---------------- NT GEMM core -----------------------------------------------
// C[M,N] = A[M,K] @ Bt[N,K]^T.  A, Bt bf16 row-major (both contiguous along k).
// EPI 0: bf16 out.  EPI 1: fp32 out = acc + residual.
// EPI 2: fp32 out = acc*scale with causal mask (col>row -> -inf), block skip.
template <int EPI>
__global__ void gemm_nt_wmma(const __bf16* __restrict__ A, int lda,
                             const __bf16* __restrict__ Bt, int ldb,
                             void* __restrict__ Cout, int ldc,
                             const float* __restrict__ residual,
                             float scale, int Kd) {
  const int m0 = blockIdx.x * BM;
  const int n0 = blockIdx.y * BN;

  if (EPI == 2 && n0 > m0 + (BM - 1)) {  // fully-masked score block
    float* Sc = (float*)Cout;
    for (int idx = threadIdx.x; idx < BM * BN; idx += 256)
      Sc[(size_t)(m0 + (idx >> 7)) * ldc + n0 + (idx & 127)] = -__builtin_inff();
    return;
  }

  // A0 | A1 | B0 | B1
  __shared__ __attribute__((aligned(16))) __bf16 smem[4 * TILE];

  const int tid  = threadIdx.x;
  const int lane = tid & 31;
  const int wid  = tid >> 5;
  const int wm   = (wid & 3) * 32;   // wave row offset in block tile
  const int wn   = (wid >> 2) * 64;  // wave col offset

  // Per-thread async staging: 2 x 32B rows (one A, one B) per K-step.
  const int lr = tid >> 1;   // 0..127 tile row
  const int lh = tid & 1;    // 16-elem half of the 32-elem k-slice
  const __bf16* gA = A  + (size_t)(m0 + lr) * lda + lh * 16;
  const __bf16* gB = Bt + (size_t)(n0 + lr) * ldb + lh * 16;
  const unsigned smem0 = (unsigned)(size_t)smem;  // low 32 bits = LDS offset
  const unsigned stg   = (unsigned)((lr * LDSPAD + lh * 16) * 2);

  v8f acc[2][4] = {};

  const int ktiles = Kd / BK;
  // prologue: tile 0 -> buffer 0
  async_copy32B(smem0 + 0 * TILEB + stg, gA);
  async_copy32B(smem0 + 2 * TILEB + stg, gB);

  for (int kt = 0; kt < ktiles; kt++) {
    const int cur = kt & 1;
    if (kt + 1 < ktiles) {
      const int k1 = (kt + 1) * BK;
      async_copy32B(smem0 + (unsigned)((cur ^ 1) * TILEB) + stg, gA + k1);
      async_copy32B(smem0 + (unsigned)((2 + (cur ^ 1)) * TILEB) + stg, gB + k1);
      wait_async_le4();        // previous (current) tile's 4 ops retired
    } else {
      wait_async_0();
    }
    __syncthreads();           // whole tile visible to all waves

    const __bf16* Ab = smem + cur * TILE;
    const __bf16* Bb = smem + (2 + cur) * TILE;
    const int fm = lane & 15, fh = lane >> 4;

    v16bf afr[2], bfr[4];
#pragma unroll
    for (int i = 0; i < 2; i++)
      afr[i] = load_frag(Ab + (wm + i * 16 + fm) * LDSPAD, fh);
#pragma unroll
    for (int j = 0; j < 4; j++)
      bfr[j] = load_frag(Bb + (wn + j * 16 + fm) * LDSPAD, fh);

#pragma unroll
    for (int i = 0; i < 2; i++)
#pragma unroll
      for (int j = 0; j < 4; j++)
        acc[i][j] = __builtin_amdgcn_wmma_f32_16x16x32_bf16(
            false, afr[i], false, bfr[j], (short)0, acc[i][j], false, false);

    __syncthreads();           // all waves done reading before buffer reuse
  }

  // Epilogue. C layout: VGPR r -> row = r + 8*(lane/16), col = lane%16.
  const int fm = lane & 15, fh = lane >> 4;
#pragma unroll
  for (int i = 0; i < 2; i++) {
#pragma unroll
    for (int j = 0; j < 4; j++) {
      const int col = n0 + wn + j * 16 + fm;
#pragma unroll
      for (int r = 0; r < 8; r++) {
        const int row = m0 + wm + i * 16 + fh * 8 + r;
        float v = acc[i][j][r];
        if (EPI == 0) {
          ((__bf16*)Cout)[(size_t)row * ldc + col] = f2bf(v);
        } else if (EPI == 1) {
          ((float*)Cout)[(size_t)row * ldc + col] =
              v + residual[(size_t)row * ldc + col];
        } else {
          v *= scale;
          if (col > row) v = -__builtin_inff();
          ((float*)Cout)[(size_t)row * ldc + col] = v;
        }
      }
    }
  }
}

// ---------------- row softmax: fp32 scores -> bf16 probabilities -------------
__global__ void softmax_row_bf16(const float* __restrict__ Sc,
                                 __bf16* __restrict__ P, int n) {
  __shared__ float red[256];
  const int row = blockIdx.x;
  const int tid = threadIdx.x;
  const float* r = Sc + (size_t)row * n;

  float v[8];
  float m = -__builtin_inff();
#pragma unroll
  for (int i = 0; i < 8; i++) { v[i] = r[tid + i * 256]; m = fmaxf(m, v[i]); }
  red[tid] = m; __syncthreads();
  for (int o = 128; o > 0; o >>= 1) {
    if (tid < o) red[tid] = fmaxf(red[tid], red[tid + o]);
    __syncthreads();
  }
  m = red[0]; __syncthreads();

  float s = 0.f;
#pragma unroll
  for (int i = 0; i < 8; i++) { v[i] = __expf(v[i] - m); s += v[i]; }  // exp(-inf)=0
  red[tid] = s; __syncthreads();
  for (int o = 128; o > 0; o >>= 1) {
    if (tid < o) red[tid] += red[tid + o];
    __syncthreads();
  }
  const float inv = 1.0f / red[0];

  __bf16* pr = P + (size_t)row * n;
#pragma unroll
  for (int i = 0; i < 8; i++) pr[tid + i * 256] = f2bf(v[i] * inv);
}

// ---------------------------------------------------------------------------
extern "C" void kernel_launch(void* const* d_in, const int* in_sizes, int n_in,
                              void* d_out, int out_size, void* d_ws, size_t ws_size,
                              hipStream_t stream) {
  (void)in_sizes; (void)n_in; (void)out_size; (void)ws_size;
  const float* x     = (const float*)d_in[0];  // [4,2048,2048]
  const float* qkv   = (const float*)d_in[1];  // [2048,6144]
  const float* oproj = (const float*)d_in[2];  // [2048,2048]
  const float* gamma = (const float*)d_in[3];  // [2048]
  const float* beta  = (const float*)d_in[4];  // [2048]
  float* out = (float*)d_out;                  // [4,2048,2048]

  // Workspace carve-up (~224 MB)
  char* ws = (char*)d_ws;
  __bf16* xn       = (__bf16*)ws; ws += (size_t)NROWS * HIDDEN * 2;  // 32MB
  __bf16* qkv_wt   = (__bf16*)ws; ws += (size_t)QKV3 * HIDDEN * 2;   // 24MB [6144][2048]
  __bf16* oproj_wt = (__bf16*)ws; ws += (size_t)HIDDEN * KEY * 2;    //  8MB [2048][2048]
  __bf16* qkvo     = (__bf16*)ws; ws += (size_t)NROWS * QKV3 * 2;    // 96MB
  float*  scores   = (float*)ws;  ws += (size_t)SEQ * SEQ * 4;       // 16MB (reused)
  __bf16* pmat     = (__bf16*)ws; ws += (size_t)SEQ * SEQ * 2;       //  8MB (reused)
  __bf16* vt       = (__bf16*)ws; ws += (size_t)KEY * SEQ * 2;       //  8MB (reused)
  __bf16* attn     = (__bf16*)ws; ws += (size_t)NROWS * KEY * 2;     // 32MB

  const float scale = 1.0f / sqrtf((float)KEY);

  // 1) weights: cast fp32 -> bf16 with transpose (so all GEMM B operands are [N][K])
  transpose_to_bf16<float><<<dim3(QKV3 / 64, HIDDEN / 64), 256, 0, stream>>>(
      qkv, QKV3, qkv_wt, HIDDEN);
  transpose_to_bf16<float><<<dim3(HIDDEN / 64, KEY / 64), 256, 0, stream>>>(
      oproj, HIDDEN, oproj_wt, KEY);

  // 2) LayerNorm -> bf16 activations
  layernorm_bf16<<<NROWS, 256, 0, stream>>>(x, gamma, beta, xn);

  // 3) QKV projection: [8192,6144] = xn @ qkv_wt^T
  gemm_nt_wmma<0><<<dim3(NROWS / BM, QKV3 / BN), 256, 0, stream>>>(
      xn, HIDDEN, qkv_wt, HIDDEN, (void*)qkvo, QKV3, nullptr, 0.f, HIDDEN);

  // 4) per-batch causal attention (scores/pmat/vt buffers reused serially)
  for (int b = 0; b < BATCH; b++) {
    const __bf16* Qb = qkvo + (size_t)b * SEQ * QKV3;
    const __bf16* Kb = Qb + KEY;
    const __bf16* Vb = Qb + 2 * KEY;

    // scores = scale * Q @ K^T with causal mask (K rows already k-contiguous)
    gemm_nt_wmma<2><<<dim3(SEQ / BM, SEQ / BN), 256, 0, stream>>>(
        Qb, QKV3, Kb, QKV3, (void*)scores, SEQ, nullptr, scale, KEY);

    softmax_row_bf16<<<SEQ, 256, 0, stream>>>(scores, pmat, SEQ);

    // V^T: [key][seq] so PV gemm is also NT-style
    transpose_to_bf16<__bf16><<<dim3(KEY / 64, SEQ / 64), 256, 0, stream>>>(
        Vb, QKV3, vt, SEQ);

    gemm_nt_wmma<0><<<dim3(SEQ / BM, KEY / BN), 256, 0, stream>>>(
        pmat, SEQ, vt, SEQ, (void*)(attn + (size_t)b * SEQ * KEY), KEY,
        nullptr, 0.f, SEQ);
  }

  // 5) o_proj + residual: fp32 out = attn @ oproj_wt^T + x
  gemm_nt_wmma<1><<<dim3(NROWS / BM, HIDDEN / BN), 256, 0, stream>>>(
      attn, KEY, oproj_wt, KEY, (void*)out, HIDDEN, x, 0.f, KEY);
}
